// GraphAttentionLayer_33878702030945
// MI455X (gfx1250) — matrix operand order repaired and checked by reading, not compiled, not run on hardware
//
#include <hip/hip_runtime.h>
#include <hip/hip_bf16.h>
#include <cstdint>

// ---------------- problem constants ----------------
#define GDIM     100000
#define IN_DIM   256
#define OUT_DIM  128
#define BQ       4096
#define NN       8192
#define ALPHA    0.2f
#define NEG_INF  -9.0e15f
#define NTILES   (BQ / 16)           // 256
#define MAXSPLIT 8

// ---------------- types ----------------
typedef __attribute__((ext_vector_type(16))) __bf16 v16bf;
typedef __attribute__((ext_vector_type(8)))  float  v8f;
typedef __attribute__((ext_vector_type(4)))  int    i4v;
typedef __attribute__((ext_vector_type(4)))  float  f4v;

struct alignas(16) Q16 { uint32_t a, b, c, d; };

union V16BF { v16bf v; Q16 q[2]; __bf16 e[16]; };
union BF8PK { __bf16 e[8]; Q16 q; };

__device__ inline v8f v8f_zero() {
  v8f z;
#pragma unroll
  for (int i = 0; i < 8; ++i) z[i] = 0.f;
  return z;
}

// ---------------- workspace layout (bytes) ----------------
#define WS_HNT   0                  // bf16 [128][8192]  2,097,152
#define WS_WA1   2097152            // f32  [256]
#define WS_WA2   2098176            // f32  [256]
#define WS_E1    2099200            // f32  [4096]
#define WS_E2    2115584            // f32  [8192]
#define WS_PART  2148352            // pm/pl/pacc, sized by chosen split

// ================= K0: wa1 = W @ a1, wa2 = W @ a2 =================
__global__ __launch_bounds__(IN_DIM) void gat_wa(const float* __restrict__ W,
                                                 const float* __restrict__ a1,
                                                 const float* __restrict__ a2,
                                                 float* __restrict__ wa1,
                                                 float* __restrict__ wa2) {
  int k = threadIdx.x;                      // 0..255
  float s1 = 0.f, s2 = 0.f;
  const float* wr = W + (size_t)k * OUT_DIM;
#pragma unroll 4
  for (int n = 0; n < OUT_DIM; ++n) {
    float w = wr[n];
    s1 = fmaf(w, a1[n], s1);
    s2 = fmaf(w, a2[n], s2);
  }
  wa1[k] = s1;
  wa2[k] = s2;
}

// ================= K1: e1[i] = gf[nodes[i]].wa1 ; e2[j] = gf[un[j]].wa2 =====
__global__ __launch_bounds__(128) void gat_escore(const float* __restrict__ gf,
                                                  const int* __restrict__ nodes,
                                                  const int* __restrict__ un,
                                                  const float* __restrict__ wa1,
                                                  const float* __restrict__ wa2,
                                                  float* __restrict__ e1,
                                                  float* __restrict__ e2) {
  int wave = threadIdx.x >> 5;
  int lane = threadIdx.x & 31;
  int row  = blockIdx.x * 4 + wave;         // 0 .. BQ+NN-1
  const float* vec;
  float* outp;
  int g;
  if (row < BQ) { g = nodes[row]; vec = wa1; outp = e1 + row; }
  else          { int j = row - BQ; g = un[j]; vec = wa2; outp = e2 + j; }
  const float* xr = gf + (size_t)g * IN_DIM;
  float s = 0.f;
#pragma unroll
  for (int c = 0; c < IN_DIM; c += 32)
    s = fmaf(__builtin_nontemporal_load(xr + c + lane), vec[c + lane], s);
#pragma unroll
  for (int off = 16; off >= 1; off >>= 1) s += __shfl_xor(s, off, 32);
  if (lane == 0) *outp = s;
}

// ================= K2: hnT[n][j] = bf16( (gf[un] @ W)[j][n] ) ==============
// 128 threads = 4 waves, each wave produces a 16(j) x 128(n) tile via
// 8x8 = 64 v_wmma_f32_16x16x32_bf16.  W staged transposed (bf16) in LDS.
__global__ __launch_bounds__(128) void gat_project(const float* __restrict__ gf,
                                                   const int* __restrict__ un,
                                                   const float* __restrict__ W,
                                                   __bf16* __restrict__ hnT) {
  __shared__ __bf16 Wt[OUT_DIM * IN_DIM];   // Wt[n][k], 64 KB
  int tid = threadIdx.x;
  for (int idx = tid; idx < IN_DIM * OUT_DIM; idx += 128) {
    int k = idx >> 7, n = idx & (OUT_DIM - 1);
    Wt[n * IN_DIM + k] = (__bf16)W[idx];
  }
  __syncthreads();

  int w = tid >> 5, lane = tid & 31;
  int m = lane & 15, h = lane >> 4;
  int j0 = blockIdx.x * 64;
  int jrow = j0 + 16 * w + m;               // gathered row this lane feeds A with
  const float* xr = gf + (size_t)un[jrow] * IN_DIM;

  v8f acc[8];
#pragma unroll
  for (int nt = 0; nt < 8; ++nt) acc[nt] = v8f_zero();

  for (int kt = 0; kt < 8; ++kt) {
    // A fragment: lane holds row m, k = 32kt + {8h..8h+7} and {16+8h..16+8h+7}
    f4v x0 = __builtin_nontemporal_load((const f4v*)(xr + 32 * kt + 8 * h));
    f4v x1 = __builtin_nontemporal_load((const f4v*)(xr + 32 * kt + 8 * h) + 1);
    f4v x2 = __builtin_nontemporal_load((const f4v*)(xr + 32 * kt + 16 + 8 * h));
    f4v x3 = __builtin_nontemporal_load((const f4v*)(xr + 32 * kt + 16 + 8 * h) + 1);
    V16BF a;
    a.e[0] = (__bf16)x0.x; a.e[1] = (__bf16)x0.y; a.e[2] = (__bf16)x0.z; a.e[3] = (__bf16)x0.w;
    a.e[4] = (__bf16)x1.x; a.e[5] = (__bf16)x1.y; a.e[6] = (__bf16)x1.z; a.e[7] = (__bf16)x1.w;
    a.e[8] = (__bf16)x2.x; a.e[9] = (__bf16)x2.y; a.e[10] = (__bf16)x2.z; a.e[11] = (__bf16)x2.w;
    a.e[12] = (__bf16)x3.x; a.e[13] = (__bf16)x3.y; a.e[14] = (__bf16)x3.z; a.e[15] = (__bf16)x3.w;
#pragma unroll
    for (int nt = 0; nt < 8; ++nt) {
      // B fragment: lane holds col n = 16nt+m, k = 32kt + {8h..8h+7} and {16+8h..16+8h+7}
      V16BF b;
      const __bf16* bp = Wt + (16 * nt + m) * IN_DIM + 32 * kt + 8 * h;
      b.q[0] = *(const Q16*)bp; b.q[1] = *(const Q16*)(bp + 16);
      acc[nt] = __builtin_amdgcn_wmma_f32_16x16x32_bf16(
          false, a.v, false, b.v, (short)0, acc[nt], false, false);
    }
  }
  // store transposed: hnT[n][j], C row r+8h -> j = j0 + 16w + 8h + r (contiguous)
#pragma unroll
  for (int nt = 0; nt < 8; ++nt) {
    BF8PK st;
#pragma unroll
    for (int r = 0; r < 8; ++r) st.e[r] = (__bf16)acc[nt][r];
    int n = 16 * nt + m;
    *(Q16*)(hnT + (size_t)n * NN + j0 + 16 * w + 8 * h) = st.q;
  }
}

// ================= K3: fused masked softmax + P @ V (flash style) ==========
// 1 wave per (16-row tile, N/split-col split).  8 WMMAs per 32-column chunk.
// mask is streamed with non-temporal loads (read-once, 134 MB); hnT/e2 stay
// cache-resident (RT) since they are re-read by every tile.
__global__ __launch_bounds__(32) void gat_attn(const int* __restrict__ mask,
                                               const float* __restrict__ e1,
                                               const float* __restrict__ e2,
                                               const __bf16* __restrict__ hnT,
                                               float* __restrict__ pm,
                                               float* __restrict__ pl,
                                               float* __restrict__ pacc,
                                               int split, int jspan) {
  int lane = threadIdx.x;
  int unit = blockIdx.x;
  int tile = unit / split, sp = unit - tile * split;
  int i0 = tile * 16, jb = sp * jspan;
  int m = lane & 15, h = lane >> 4;

  float e1r = e1[i0 + m];
  const int* mrow = mask + (size_t)(i0 + m) * NN;

  float m_run = -3.0e38f, l_run = 0.f;
  v8f acc[8];
#pragma unroll
  for (int nt = 0; nt < 8; ++nt) acc[nt] = v8f_zero();

  for (int jc = 0; jc < jspan; jc += 32) {
    int j0 = jb + jc;
    // lane covers j = j0+8h+{0..7} (k=8h+t) and j0+16+8h+{0..7} (k=16+8h+t)
    const f4v* ep = (const f4v*)(e2 + j0 + 8 * h);
    f4v ea0 = ep[0], ea1 = ep[1];
    const f4v* eq = (const f4v*)(e2 + j0 + 16 + 8 * h);
    f4v eb0 = eq[0], eb1 = eq[1];
    const i4v* mp = (const i4v*)(mrow + j0 + 8 * h);
    i4v ma0 = __builtin_nontemporal_load(mp);
    i4v ma1 = __builtin_nontemporal_load(mp + 1);
    const i4v* mq = (const i4v*)(mrow + j0 + 16 + 8 * h);
    i4v mb0 = __builtin_nontemporal_load(mq);
    i4v mb1 = __builtin_nontemporal_load(mq + 1);

    float ev[16];
#define GAT_EVAL(d, ee, mm)                                        \
    { float tt = e1r + (ee); tt = tt >= 0.f ? tt : ALPHA * tt;     \
      ev[d] = (mm) > 0 ? tt : NEG_INF; }
    GAT_EVAL(0,  ea0.x, ma0.x) GAT_EVAL(1,  ea0.y, ma0.y)
    GAT_EVAL(2,  ea0.z, ma0.z) GAT_EVAL(3,  ea0.w, ma0.w)
    GAT_EVAL(4,  ea1.x, ma1.x) GAT_EVAL(5,  ea1.y, ma1.y)
    GAT_EVAL(6,  ea1.z, ma1.z) GAT_EVAL(7,  ea1.w, ma1.w)
    GAT_EVAL(8,  eb0.x, mb0.x) GAT_EVAL(9,  eb0.y, mb0.y)
    GAT_EVAL(10, eb0.z, mb0.z) GAT_EVAL(11, eb0.w, mb0.w)
    GAT_EVAL(12, eb1.x, mb1.x) GAT_EVAL(13, eb1.y, mb1.y)
    GAT_EVAL(14, eb1.z, mb1.z) GAT_EVAL(15, eb1.w, mb1.w)
#undef GAT_EVAL

    float cmax = ev[0];
#pragma unroll
    for (int t = 1; t < 16; ++t) cmax = fmaxf(cmax, ev[t]);
    cmax = fmaxf(cmax, __shfl_xor(cmax, 16, 32));     // both halves of row m

    float m_new = fmaxf(m_run, cmax);
    float scale = __expf(m_run - m_new);              // 1.0 if max unchanged
    if (__any(scale < 1.f)) {
#pragma unroll
      for (int r = 0; r < 8; ++r) {
        float sr = __shfl(scale, r + 8 * h, 32);      // scale of row r+8h
#pragma unroll
        for (int nt = 0; nt < 8; ++nt) acc[nt][r] *= sr;
      }
    }
    l_run *= scale;
    m_run = m_new;

    V16BF a;
    float psum = 0.f;
#pragma unroll
    for (int t = 0; t < 16; ++t) {
      // masked lanes carry ev == NEG_INF; force exact 0 probability
      float p = (ev[t] > -1.0e14f) ? __expf(ev[t] - m_run) : 0.f;
      psum += p;
      a.e[t] = (__bf16)p;
    }
    l_run += psum + __shfl_xor(psum, 16, 32);

#pragma unroll
    for (int nt = 0; nt < 8; ++nt) {
      // V fragment: col n = 16nt+m, k(=j-offset) = 8h + {0..7} and 16 + 8h + {0..7} in hnT
      V16BF b;
      const __bf16* bp = hnT + (size_t)(16 * nt + m) * NN + j0 + 8 * h;
      b.q[0] = *(const Q16*)bp; b.q[1] = *(const Q16*)(bp + 16);
      acc[nt] = __builtin_amdgcn_wmma_f32_16x16x32_bf16(
          false, a.v, false, b.v, (short)0, acc[nt], false, false);
    }
  }

  if (h == 0) {
    pm[unit * 16 + m] = m_run;
    pl[unit * 16 + m] = l_run;
  }
#pragma unroll
  for (int nt = 0; nt < 8; ++nt)
#pragma unroll
    for (int r = 0; r < 8; ++r)
      __builtin_nontemporal_store(
          acc[nt][r],
          pacc + ((size_t)unit * 16 + r + 8 * h) * OUT_DIM + 16 * nt + m);
}

// ================= K4: combine splits + softmax/L2 normalize ===============
__global__ __launch_bounds__(128) void gat_combine(const float* __restrict__ pm,
                                                   const float* __restrict__ pl,
                                                   const float* __restrict__ pacc,
                                                   float* __restrict__ out,
                                                   int split) {
  int tile = blockIdx.x, tid = threadIdx.x;
  __shared__ float cs[MAXSPLIT][16];
  __shared__ float linv[16];
  __shared__ float vrow[16][OUT_DIM];
  __shared__ float rnorm[16];

  if (tid < 16) {
    float M = -3.0e38f;
    for (int s = 0; s < split; ++s) M = fmaxf(M, pm[(tile * split + s) * 16 + tid]);
    float L = 0.f;
    for (int s = 0; s < split; ++s) {
      float c = __expf(pm[(tile * split + s) * 16 + tid] - M);
      cs[s][tid] = c;
      L += pl[(tile * split + s) * 16 + tid] * c;
    }
    linv[tid] = 1.f / fmaxf(L, 1e-30f);
  }
  __syncthreads();

  int n = tid;                               // 0..127
  for (int i = 0; i < 16; ++i) {
    float v = 0.f;
    for (int s = 0; s < split; ++s)
      v += __builtin_nontemporal_load(
               pacc + (((size_t)(tile * split + s)) * 16 + i) * OUT_DIM + n) * cs[s][i];
    vrow[i][n] = v * linv[i];
  }
  __syncthreads();

  if (tid < 16) {
    float ss = 0.f;
#pragma unroll 4
    for (int c = 0; c < OUT_DIM; ++c) { float v = vrow[tid][c]; ss = fmaf(v, v, ss); }
    rnorm[tid] = 1.f / fmaxf(sqrtf(ss), 1e-12f);
  }
  __syncthreads();

  for (int i = 0; i < 16; ++i)
    __builtin_nontemporal_store(vrow[i][n] * rnorm[i],
                                out + ((size_t)(tile * 16 + i)) * OUT_DIM + n);
}

// ================= host launcher ===========================================
extern "C" void kernel_launch(void* const* d_in, const int* in_sizes, int n_in,
                              void* d_out, int out_size, void* d_ws, size_t ws_size,
                              hipStream_t stream) {
  (void)in_sizes; (void)n_in; (void)out_size;
  const float* gf    = (const float*)d_in[0];
  const int*   nodes = (const int*)d_in[1];
  const int*   un    = (const int*)d_in[2];
  const int*   mask  = (const int*)d_in[3];
  const float* W     = (const float*)d_in[4];
  const float* a1    = (const float*)d_in[5];
  const float* a2    = (const float*)d_in[6];
  float* out = (float*)d_out;

  // pick the largest N-split (parallelism for the flash kernel) that fits ws.
  // per-unit partials: pm 16*4 + pl 16*4 + pacc 16*128*4 = 8320 bytes.
  int split = MAXSPLIT;
  while (split > 1 &&
         (size_t)WS_PART + (size_t)NTILES * split * 8320u > ws_size)
    split >>= 1;
  int units = NTILES * split;
  int jspan = NN / split;

  char* ws = (char*)d_ws;
  __bf16* hnT = (__bf16*)(ws + WS_HNT);
  float*  wa1 = (float*)(ws + WS_WA1);
  float*  wa2 = (float*)(ws + WS_WA2);
  float*  e1  = (float*)(ws + WS_E1);
  float*  e2  = (float*)(ws + WS_E2);
  float*  pm  = (float*)(ws + WS_PART);
  float*  pl  = pm + (size_t)units * 16;
  float*  pacc= pl + (size_t)units * 16;

  gat_wa<<<dim3(1), dim3(IN_DIM), 0, stream>>>(W, a1, a2, wa1, wa2);
  gat_escore<<<dim3((BQ + NN) / 4), dim3(128), 0, stream>>>(gf, nodes, un, wa1, wa2, e1, e2);
  gat_project<<<dim3(NN / 64), dim3(128), 0, stream>>>(gf, un, W, hnT);
  gat_attn<<<dim3(units), dim3(32), 0, stream>>>(mask, e1, e2, hnT, pm, pl, pacc, split, jspan);
  gat_combine<<<dim3(NTILES), dim3(128), 0, stream>>>(pm, pl, pacc, out, split);
}
